// GATClassifier_9758165697215
// MI455X (gfx1250) — matrix operand order repaired and checked by reading, not compiled, not run on hardware
//
#include <hip/hip_runtime.h>
#include <hip/hip_bf16.h>

// ---------------------------------------------------------------------------
// GAT classifier for MI455X (gfx1250). Dense projections run on bf16 WMMA
// (v_wmma_f32_16x16x32_bf16, fp32 accumulate) with a 16x64 per-wave tile
// (4 accumulators sharing one A fragment). N=50000 is a multiple of 16 and
// M in {256,64} a multiple of 64, so the GEMM has zero divergence: EXEC is
// all-ones at every WMMA. Attention/softmax/scatter stay in fp32.
// ---------------------------------------------------------------------------

#define NN   50000      // multiple of 16
#define EE   800000
#define ETOT (EE + NN)
#define INC  128
#define HIDC 64
#define HEADS 4
#define NCL  200
#define GG   64
#define NEG_SLOPE 0.2f

typedef __attribute__((ext_vector_type(16))) __bf16 v16bf;
typedef __attribute__((ext_vector_type(8)))  __bf16 v8bf;
typedef __attribute__((ext_vector_type(8)))  float  v8f;

__device__ __forceinline__ __bf16 f2bf(float f) {
  union { float f; unsigned u; } v; v.f = f;
  unsigned r = v.u + 0x7FFFu + ((v.u >> 16) & 1u);   // round-to-nearest-even
  unsigned short h = (unsigned short)(r >> 16);
  return __builtin_bit_cast(__bf16, h);
}

__device__ __forceinline__ void atomicMaxF(float* addr, float val) {
  if (val >= 0.0f)
    atomicMax((int*)addr, __float_as_int(val));
  else
    atomicMin((unsigned int*)addr, __float_as_uint(val));
}

// ---------------- elementwise helpers ----------------
__global__ void k_fill(float* __restrict__ p, float v, int n) {
  int i = blockIdx.x * blockDim.x + threadIdx.x;
  if (i < n) p[i] = v;
}

__global__ void k_cast_bf16(const float* __restrict__ in, __bf16* __restrict__ out, int n) {
  int i = blockIdx.x * blockDim.x + threadIdx.x;
  if (i < n) out[i] = f2bf(in[i]);
}

// W [K,M] fp32 -> WT [M,K] bf16
__global__ void k_transpose_w(const float* __restrict__ W, __bf16* __restrict__ WT, int K, int M) {
  int i = blockIdx.x * blockDim.x + threadIdx.x;
  if (i >= K * M) return;
  int k = i / M, m = i - k * M;
  WT[(size_t)m * K + k] = f2bf(W[i]);
}

// ---------------- WMMA GEMM: D[n,m] = A[n,k](bf16) @ BT[m,k]^T(bf16) ----------
// One wave -> one 16x64 tile (4 accumulators share the A fragment).
// Requires n % 16 == 0, m % 64 == 0, k % 32 == 0 (true for all call sites).
__global__ void k_gemm_bf16(const __bf16* __restrict__ A, const __bf16* __restrict__ BT,
                            float* __restrict__ D, int n, int k, int m) {
  int wid  = (blockIdx.x * blockDim.x + threadIdx.x) >> 5;
  int lane = threadIdx.x & 31;
  int tilesM = m >> 6;                          // 64-wide column groups
  int tr = wid / tilesM, tc = wid - tr * tilesM;
  if (tr >= (n >> 4)) return;                   // wave-uniform exit
  int rowBase = tr << 4, colBase = tc << 6;
  int half = lane >> 4, l16 = lane & 15;

  // A: lane l16 holds row rowBase+l16; 16 contiguous K per half
  const __bf16* Ap = A + (size_t)(rowBase + l16) * k + half * 16;
  // B (pre-transposed [M,K]): lane l16 holds column; 16 contiguous K per half
  const __bf16* Bp = BT + (size_t)(colBase + l16) * k + half * 16;
  const size_t bstride = (size_t)16 * k;        // next 16-column tile

  v8f acc0 = {}, acc1 = {}, acc2 = {}, acc3 = {};
  for (int kk = 0; kk < k; kk += 32) {
    v8bf lo = *(const v8bf*)(Ap + kk);
    v8bf hi = *(const v8bf*)(Ap + kk + 8);
    v16bf a;
#pragma unroll
    for (int i = 0; i < 8; ++i) { a[i] = lo[i]; a[8 + i] = hi[i]; }
    v16bf b0 = *(const v16bf*)(Bp + kk);
    v16bf b1 = *(const v16bf*)(Bp + bstride + kk);
    v16bf b2 = *(const v16bf*)(Bp + 2 * bstride + kk);
    v16bf b3 = *(const v16bf*)(Bp + 3 * bstride + kk);
    acc0 = __builtin_amdgcn_wmma_f32_16x16x32_bf16(false, a, false, b0, (short)0, acc0, false, false);
    acc1 = __builtin_amdgcn_wmma_f32_16x16x32_bf16(false, a, false, b1, (short)0, acc1, false, false);
    acc2 = __builtin_amdgcn_wmma_f32_16x16x32_bf16(false, a, false, b2, (short)0, acc2, false, false);
    acc3 = __builtin_amdgcn_wmma_f32_16x16x32_bf16(false, a, false, b3, (short)0, acc3, false, false);
  }

  // D layout: lane half 0 -> rows 0..7, half 1 -> rows 8..15; col = l16
  float* Dp = D + (size_t)(rowBase + 8 * half) * m + colBase + l16;
#pragma unroll
  for (int i = 0; i < 8; ++i) {
    Dp[(size_t)i * m +  0] = acc0[i];
    Dp[(size_t)i * m + 16] = acc1[i];
    Dp[(size_t)i * m + 32] = acc2[i];
    Dp[(size_t)i * m + 48] = acc3[i];
  }
}

// ---------------- attention coefficients per node ----------------
__global__ void k_alpha(const float* __restrict__ Hm, const float* __restrict__ as,
                        const float* __restrict__ ad, float* __restrict__ aS,
                        float* __restrict__ aD, int n, int H, int C) {
  int idx = blockIdx.x * blockDim.x + threadIdx.x;
  if (idx >= n * H) return;
  int node = idx / H, h = idx - node * H;
  const float* hp  = Hm + (size_t)node * H * C + h * C;
  const float* asp = as + h * C;
  const float* adp = ad + h * C;
  float s = 0.f, d = 0.f;
#pragma unroll 4
  for (int c = 0; c < C; ++c) { float v = hp[c]; s += v * asp[c]; d += v * adp[c]; }
  aS[idx] = s; aD[idx] = d;
}

// ---------------- edge passes ----------------
__global__ void k_edge_max(const int* __restrict__ src, const int* __restrict__ dst,
                           const float* __restrict__ aS, const float* __restrict__ aD,
                           float* __restrict__ emax, int H) {
  int idx = blockIdx.x * blockDim.x + threadIdx.x;
  if (idx >= ETOT * H) return;
  int e = idx / H, h = idx - e * H;
  int s = e < EE ? src[e] : e - EE;
  int d = e < EE ? dst[e] : e - EE;
  float v = aS[s * H + h] + aD[d * H + h];
  v = v > 0.f ? v : NEG_SLOPE * v;
  atomicMaxF(&emax[d * H + h], v);
}

__global__ void k_edge_exp(const int* __restrict__ src, const int* __restrict__ dst,
                           const float* __restrict__ aS, const float* __restrict__ aD,
                           const float* __restrict__ emax, float* __restrict__ eexp,
                           float* __restrict__ denom, int H) {
  int idx = blockIdx.x * blockDim.x + threadIdx.x;
  if (idx >= ETOT * H) return;
  int e = idx / H, h = idx - e * H;
  int s = e < EE ? src[e] : e - EE;
  int d = e < EE ? dst[e] : e - EE;
  float v = aS[s * H + h] + aD[d * H + h];
  v = v > 0.f ? v : NEG_SLOPE * v;
  float ex = __expf(v - emax[d * H + h]);
  eexp[idx] = ex;
  atomicAdd(&denom[d * H + h], ex);
}

// one wave per (edge, head); lanes cover channels
__global__ void k_edge_agg(const int* __restrict__ src, const int* __restrict__ dst,
                           const float* __restrict__ Hm, const float* __restrict__ eexp,
                           const float* __restrict__ denom, float* __restrict__ out,
                           int H, int C) {
  int gt = blockIdx.x * blockDim.x + threadIdx.x;
  int wid = gt >> 5, lane = gt & 31;
  if (wid >= ETOT * H) return;
  int e = wid / H, h = wid - e * H;
  int s = e < EE ? src[e] : e - EE;
  int d = e < EE ? dst[e] : e - EE;
  float alpha = eexp[wid] / (denom[d * H + h] + 1e-16f);
  const float* hs = Hm + (size_t)s * H * C + h * C;
  float* po = out + (size_t)d * H * C + h * C;
  for (int c = lane; c < C; c += 32)
    atomicAdd(&po[c], hs[c] * alpha);
}

__global__ void k_bias_elu(float* __restrict__ h, const float* __restrict__ b, int n, int m) {
  int i = blockIdx.x * blockDim.x + threadIdx.x;
  if (i >= n * m) return;
  float v = h[i] + b[i % m];
  h[i] = v > 0.f ? v : (__expf(v) - 1.f);
}

// ---------------- pooling + classifier ----------------
__global__ void k_pool_scatter(const float* __restrict__ h, const int* __restrict__ batch,
                               float* __restrict__ sums, float* __restrict__ cnt) {
  int i = blockIdx.x * blockDim.x + threadIdx.x;
  if (i >= NN * HIDC) return;
  int node = i >> 6, c = i & 63;
  int g = batch[node];
  atomicAdd(&sums[g * HIDC + c], h[i]);
  if (c == 0) atomicAdd(&cnt[g], 1.0f);
}

__global__ void k_pool_div(float* __restrict__ sums, const float* __restrict__ cnt) {
  int i = blockIdx.x * blockDim.x + threadIdx.x;
  if (i >= GG * HIDC) return;
  sums[i] /= fmaxf(cnt[i >> 6], 1.0f);
}

__global__ void k_classifier(const float* __restrict__ pooled, const float* __restrict__ Wc,
                             const float* __restrict__ bc, float* __restrict__ out) {
  int i = blockIdx.x * blockDim.x + threadIdx.x;
  if (i >= GG * NCL) return;
  int g = i / NCL, j = i - g * NCL;
  float acc = bc[j];
  const float* p = pooled + g * HIDC;
#pragma unroll 4
  for (int c = 0; c < HIDC; ++c) acc += p[c] * Wc[c * NCL + j];
  out[i] = acc;
}

// ---------------------------------------------------------------------------
static inline int cdiv(long long a, long long b) { return (int)((a + b - 1) / b); }

extern "C" void kernel_launch(void* const* d_in, const int* in_sizes, int n_in,
                              void* d_out, int out_size, void* d_ws, size_t ws_size,
                              hipStream_t stream) {
  (void)in_sizes; (void)n_in; (void)out_size; (void)ws_size;
  const float* x     = (const float*)d_in[0];
  const int*   ei    = (const int*)d_in[1];
  const int*   batch = (const int*)d_in[2];
  const float* W[3]  = { (const float*)d_in[3], (const float*)d_in[7], (const float*)d_in[11] };
  const float* AS[3] = { (const float*)d_in[4], (const float*)d_in[8], (const float*)d_in[12] };
  const float* AD[3] = { (const float*)d_in[5], (const float*)d_in[9], (const float*)d_in[13] };
  const float* B[3]  = { (const float*)d_in[6], (const float*)d_in[10], (const float*)d_in[14] };
  const float* Wc = (const float*)d_in[15];
  const float* bc = (const float*)d_in[16];
  const int* srcA = ei;
  const int* dstA = ei + EE;
  float* out = (float*)d_out;

  // workspace carve-up (256B aligned slices)
  char* p = (char*)d_ws;
  size_t off = 0;
  auto take = [&](size_t bytes) { char* q = p + off; off = (off + bytes + 255) & ~(size_t)255; return q; };
  float*  hA   = (float*)take((size_t)NN * 256 * 4);   // aggregation target / activation
  float*  hB   = (float*)take((size_t)NN * 256 * 4);   // GEMM output (pre-softmax features)
  __bf16* xbf  = (__bf16*)take((size_t)NN * 256 * 2);  // bf16 activations (GEMM A)
  __bf16* wt   = (__bf16*)take((size_t)256 * 256 * 2); // bf16 W^T (GEMM B)
  float*  aS   = (float*)take((size_t)NN * HEADS * 4);
  float*  aD   = (float*)take((size_t)NN * HEADS * 4);
  float*  emax = (float*)take((size_t)NN * HEADS * 4);
  float*  den  = (float*)take((size_t)NN * HEADS * 4);
  float*  eexp = (float*)take((size_t)ETOT * HEADS * 4);
  float*  psum = (float*)take((size_t)GG * HIDC * 4);
  float*  pcnt = (float*)take((size_t)GG * 4);

  const int T = 256;
  const int Kdim[3] = { INC, HEADS * HIDC, HEADS * HIDC };
  const int Mdim[3] = { HEADS * HIDC, HEADS * HIDC, HIDC };
  const int Hh[3]   = { HEADS, HEADS, 1 };

  for (int L = 0; L < 3; ++L) {
    int K = Kdim[L], M = Mdim[L], H = Hh[L], C = HIDC;
    const float* actIn = (L == 0) ? x : hA;

    // bf16 activation + transposed bf16 weight
    k_cast_bf16<<<cdiv((long long)NN * K, T), T, 0, stream>>>(actIn, xbf, NN * K);
    k_transpose_w<<<cdiv((long long)K * M, T), T, 0, stream>>>(W[L], wt, K, M);

    // h = act @ W  (bf16 WMMA, fp32 accumulate); one wave per 16x64 tile
    long long waves = (long long)(NN / 16) * (M / 64);
    k_gemm_bf16<<<cdiv(waves, 8), T, 0, stream>>>(xbf, wt, hB, NN, K, M);

    // per-node attention coefficients
    k_alpha<<<cdiv((long long)NN * H, T), T, 0, stream>>>(hB, AS[L], AD[L], aS, aD, NN, H, C);

    // segment softmax + weighted scatter aggregation
    k_fill<<<cdiv((long long)NN * H, T), T, 0, stream>>>(emax, -1e30f, NN * H);
    k_fill<<<cdiv((long long)NN * H, T), T, 0, stream>>>(den, 0.f, NN * H);
    k_fill<<<cdiv((long long)NN * M, T), T, 0, stream>>>(hA, 0.f, NN * M);
    k_edge_max<<<cdiv((long long)ETOT * H, T), T, 0, stream>>>(srcA, dstA, aS, aD, emax, H);
    k_edge_exp<<<cdiv((long long)ETOT * H, T), T, 0, stream>>>(srcA, dstA, aS, aD, emax, eexp, den, H);
    k_edge_agg<<<cdiv((long long)ETOT * H * 32, T), T, 0, stream>>>(srcA, dstA, hB, eexp, den, hA, H, C);

    // bias + ELU
    k_bias_elu<<<cdiv((long long)NN * M, T), T, 0, stream>>>(hA, B[L], NN, M);
  }

  // global mean pool + classifier
  k_fill<<<cdiv(GG * HIDC, T), T, 0, stream>>>(psum, 0.f, GG * HIDC);
  k_fill<<<1, GG, 0, stream>>>(pcnt, 0.f, GG);
  k_pool_scatter<<<cdiv((long long)NN * HIDC, T), T, 0, stream>>>(hA, batch, psum, pcnt);
  k_pool_div<<<cdiv(GG * HIDC, T), T, 0, stream>>>(psum, pcnt);
  k_classifier<<<cdiv(GG * NCL, T), T, 0, stream>>>(psum, Wc, bc, out);
}
